// TransformerBlock_79877801771229
// MI455X (gfx1250) — matrix operand order repaired and checked by reading, not compiled, not run on hardware
//
#include <hip/hip_runtime.h>
#include <hip/hip_bf16.h>

typedef __bf16 bf16;
typedef __attribute__((ext_vector_type(16))) __bf16 v16bf;
typedef __attribute__((ext_vector_type(8)))  float  v8f;

// Problem dims (fixed by reference)
constexpr int H_  = 16;
constexpr int K_  = 64;
constexpr int D_  = 1024;
constexpr int M_  = 4096;
constexpr int S_  = 2048;
constexpr int B_  = 4;
constexpr int NR_ = S_ * B_;          // 8192 token rows
constexpr long HK_ = (long)H_ * K_;   // 1024

// ---------------------------------------------------------------------------
// WMMA helpers (layouts per CDNA5 ISA 7.12.2, wave32)
// ---------------------------------------------------------------------------
__device__ __forceinline__ v8f wmma_bf16(v16bf a, v16bf b, v8f c) {
  return __builtin_amdgcn_wmma_f32_16x16x32_bf16(false, a, false, b, (short)0, c,
                                                 false, false);
}

// A-matrix 16x32 bf16 fragment. Row-major source, K contiguous (stride 1),
// row stride = rowStride. lane<16: row=lane, K in {0..7,16..23}; lane>=16: +8.
__device__ __forceinline__ v16bf load_a_frag(const bf16* base, long rowStride,
                                             int row0, int k0) {
  int lane = threadIdx.x & 31;
  const bf16* p = base + (long)(row0 + (lane & 15)) * rowStride + k0 + ((lane >> 4) << 3);
  v16bf a;
#pragma unroll
  for (int j = 0; j < 8; ++j) {
    int kk = ((j & 3) << 1) + ((j >> 2) << 4);
    a[2 * j]     = p[kk];
    a[2 * j + 1] = p[kk + 1];
  }
  return a;
}

// B-matrix 32x16 bf16 fragment (K x N). lane<16: col=lane, K in {0..7,16..23};
// lane>=16: col=lane-16, K in {8..15,24..31}. Generic strides so it serves both
// weight-NT GEMMs (nStride=ld, kStride=1) and the V operand in attention
// (nStride=1, kStride=token stride).
__device__ __forceinline__ v16bf load_b_frag(const bf16* base, long nStride,
                                             long kStride, int n0, int k0) {
  int lane = threadIdx.x & 31;
  const bf16* p = base + (long)(n0 + (lane & 15)) * nStride
                       + (long)(k0 + ((lane >> 4) << 3)) * kStride;
  v16bf b;
#pragma unroll
  for (int j = 0; j < 8; ++j) {
    b[2 * j]     = p[(long)(2 * j + ((j >> 2) << 3)) * kStride];
    b[2 * j + 1] = p[(long)(2 * j + 1 + ((j >> 2) << 3)) * kStride];
  }
  return b;
}

// ---------------------------------------------------------------------------
// fp32 -> bf16 conversion
// ---------------------------------------------------------------------------
__global__ void cvt_bf16_kernel(const float* __restrict__ in, bf16* __restrict__ out,
                                long n) {
  long i = (long)blockIdx.x * blockDim.x + threadIdx.x;
  if (i < n) out[i] = (bf16)in[i];
}

// ---------------------------------------------------------------------------
// NT GEMM: C[Mr x Nr] = A[Mr x Kd] * Bw[Nr x Kd]^T   (both bf16 row-major)
// Wave tile: 32(M) x 64(N). Fused bias / relu / fp32-or-bf16 store.
// ---------------------------------------------------------------------------
__global__ __launch_bounds__(256) void gemm_wmma_nt(
    const bf16* __restrict__ A, const bf16* __restrict__ Bw,
    const float* __restrict__ bias, float* __restrict__ Cf,
    bf16* __restrict__ Cb, int Mr, int Nr, int Kd, int relu) {
  int wid    = (int)((blockIdx.x * blockDim.x + threadIdx.x) >> 5);
  int tilesN = Nr >> 6;
  int rBase  = (wid / tilesN) << 5;
  int cBase  = (wid % tilesN) << 6;

  v8f acc[2][4];
#pragma unroll
  for (int i = 0; i < 2; ++i)
#pragma unroll
    for (int t = 0; t < 4; ++t) acc[i][t] = (v8f){};

  for (int k0 = 0; k0 < Kd; k0 += 32) {
    v16bf a0 = load_a_frag(A, Kd, rBase, k0);
    v16bf a1 = load_a_frag(A, Kd, rBase + 16, k0);
#pragma unroll
    for (int t = 0; t < 4; ++t) {
      v16bf b = load_b_frag(Bw, Kd, 1, cBase + 16 * t, k0);
      acc[0][t] = wmma_bf16(a0, b, acc[0][t]);
      acc[1][t] = wmma_bf16(a1, b, acc[1][t]);
    }
  }

  int lane = threadIdx.x & 31;
  int cOff = lane & 15;
  int rOff = (lane >> 4) << 3;
#pragma unroll
  for (int i = 0; i < 2; ++i)
#pragma unroll
    for (int t = 0; t < 4; ++t)
#pragma unroll
      for (int r = 0; r < 8; ++r) {
        long row = rBase + 16 * i + rOff + r;
        long col = cBase + 16 * t + cOff;
        float v = acc[i][t][r];
        if (bias) v += bias[col];
        if (relu) v = fmaxf(v, 0.0f);
        if (Cf) Cf[row * Nr + col] = v;
        else    Cb[row * Nr + col] = (bf16)v;
      }
}

// ---------------------------------------------------------------------------
// Flash attention. One wave = one (b, h, 16-query tile). Streams keys in
// blocks of 32 with online softmax; P transposed via 1KB/wave LDS so P.V is a
// clean 16x16x32 WMMA. q/k/v laid out [token(S*B), h*64+k] bf16.
// ---------------------------------------------------------------------------
__global__ __launch_bounds__(256) void flash_attn_kernel(
    const bf16* __restrict__ qb, const bf16* __restrict__ kb,
    const bf16* __restrict__ vb, const float* __restrict__ mask,
    bf16* __restrict__ avb) {
  __shared__ bf16 Plds[8][16 * 32];

  int wid  = threadIdx.x >> 5;
  int lane = threadIdx.x & 31;
  int idx  = blockIdx.x * 8 + wid;      // 0..8191
  int qt   = idx & 127;                 // 128 query tiles
  int bh   = idx >> 7;                  // 64 (b,h) pairs
  int b    = bh >> 4;
  int h    = bh & 15;

  const long tokStride = HK_ * B_;      // 4096: stride between s and s+1
  const bf16* qbase = qb + (long)b * HK_ + (long)h * K_;
  const bf16* kbase = kb + (long)b * HK_ + (long)h * K_;
  const bf16* vbase = vb + (long)b * HK_ + (long)h * K_;

  v16bf qf0 = load_a_frag(qbase, tokStride, qt * 16, 0);
  v16bf qf1 = load_a_frag(qbase, tokStride, qt * 16, 32);

  float m[8], l[8];
  v8f acc[4];
#pragma unroll
  for (int t = 0; t < 4; ++t) acc[t] = (v8f){};
#pragma unroll
  for (int r = 0; r < 8; ++r) { m[r] = -1e30f; l[r] = 0.0f; }

  int rOff  = (lane >> 4) << 3;
  int qrow0 = qt * 16 + rOff;

  for (int j0 = 0; j0 < S_; j0 += 32) {
    // scores for 32 keys (two 16x16 tiles), K-dim = 64 head dims
    v8f s0 = (v8f){}, s1 = (v8f){};
    {
      v16bf kf;
      kf = load_b_frag(kbase, tokStride, 1, j0, 0);       s0 = wmma_bf16(qf0, kf, s0);
      kf = load_b_frag(kbase, tokStride, 1, j0, 32);      s0 = wmma_bf16(qf1, kf, s0);
      kf = load_b_frag(kbase, tokStride, 1, j0 + 16, 0);  s1 = wmma_bf16(qf0, kf, s1);
      kf = load_b_frag(kbase, tokStride, 1, j0 + 16, 32); s1 = wmma_bf16(qf1, kf, s1);
    }
    int key = j0 + (lane & 15);
#pragma unroll
    for (int r = 0; r < 8; ++r) {
      long qr = qrow0 + r;
      s0[r] = s0[r] * 0.125f + mask[qr * S_ + key];
      s1[r] = s1[r] * 0.125f + mask[qr * S_ + key + 16];
    }
    // online softmax update (row stats live across the 16 lanes of a half)
#pragma unroll
    for (int r = 0; r < 8; ++r) {
      float tmax = fmaxf(s0[r], s1[r]);
#pragma unroll
      for (int o = 1; o < 16; o <<= 1) tmax = fmaxf(tmax, __shfl_xor(tmax, o, 32));
      float mn = fmaxf(m[r], tmax);
      float sc = __expf(m[r] - mn);
      float p0 = __expf(s0[r] - mn);
      float p1 = __expf(s1[r] - mn);
      s0[r] = p0; s1[r] = p1;
      float rs = p0 + p1;
#pragma unroll
      for (int o = 1; o < 16; o <<= 1) rs += __shfl_xor(rs, o, 32);
      l[r] = l[r] * sc + rs;
      m[r] = mn;
#pragma unroll
      for (int t = 0; t < 4; ++t) acc[t][r] *= sc;
    }
    // stage P (16x32) in LDS: score layout -> A-fragment layout
    bf16* P = Plds[wid];
    int pcol = lane & 15;
#pragma unroll
    for (int r = 0; r < 8; ++r) {
      P[(rOff + r) * 32 + pcol]      = (bf16)s0[r];
      P[(rOff + r) * 32 + pcol + 16] = (bf16)s1[r];
    }
    __syncthreads();
    v16bf pf = load_a_frag(P, 32, 0, 0);
    __syncthreads();
    // acc += P(16x32) * V(32x64)
#pragma unroll
    for (int t = 0; t < 4; ++t) {
      v16bf vf = load_b_frag(vbase + (long)j0 * tokStride, 1, tokStride, t * 16, 0);
      acc[t] = wmma_bf16(pf, vf, acc[t]);
    }
  }

  // normalize and store av[token, h*64+dim] as bf16
  int cOff = lane & 15;
#pragma unroll
  for (int t = 0; t < 4; ++t)
#pragma unroll
    for (int r = 0; r < 8; ++r) {
      long qr = qrow0 + r;
      float o = acc[t][r] / l[r];
      avb[(qr * B_ + b) * HK_ + (long)h * K_ + t * 16 + cOff] = (bf16)o;
    }
}

// ---------------------------------------------------------------------------
// LayerNorm over D=1024: out = (a+res - mean)/sqrt(var+eps)*g + beta
// One 256-thread block per row. Optional fp32 and bf16 outputs.
// ---------------------------------------------------------------------------
__global__ __launch_bounds__(256) void ln_kernel(
    const float* __restrict__ a, const float* __restrict__ res,
    const float* __restrict__ g, const float* __restrict__ be,
    float* __restrict__ outF, bf16* __restrict__ outB) {
  long row = blockIdx.x;
  int tid = threadIdx.x;
  float vals[4];
  float s = 0.0f, s2 = 0.0f;
#pragma unroll
  for (int i = 0; i < 4; ++i) {
    int c = tid + i * 256;
    float v = a[row * D_ + c] + res[row * D_ + c];
    vals[i] = v; s += v; s2 += v * v;
  }
#pragma unroll
  for (int o = 1; o < 32; o <<= 1) { s += __shfl_xor(s, o, 32); s2 += __shfl_xor(s2, o, 32); }
  __shared__ float rs[8], rq[8];
  if ((tid & 31) == 0) { rs[tid >> 5] = s; rq[tid >> 5] = s2; }
  __syncthreads();
  float ts = 0.0f, tq = 0.0f;
#pragma unroll
  for (int w = 0; w < 8; ++w) { ts += rs[w]; tq += rq[w]; }
  float mean = ts * (1.0f / D_);
  float var  = tq * (1.0f / D_) - mean * mean;
  float inv  = rsqrtf(var + 1e-5f);
#pragma unroll
  for (int i = 0; i < 4; ++i) {
    int c = tid + i * 256;
    float o = (vals[i] - mean) * inv * g[c] + be[c];
    if (outF) outF[row * D_ + c] = o;
    if (outB) outB[row * D_ + c] = (bf16)o;
  }
}

// ---------------------------------------------------------------------------
// Host-side orchestration
// ---------------------------------------------------------------------------
extern "C" void kernel_launch(void* const* d_in, const int* in_sizes, int n_in,
                              void* d_out, int out_size, void* d_ws, size_t ws_size,
                              hipStream_t stream) {
  (void)in_sizes; (void)n_in; (void)out_size; (void)ws_size;
  const float* x    = (const float*)d_in[0];
  const float* mask = (const float*)d_in[1];
  const float* wq   = (const float*)d_in[2];
  const float* wk   = (const float*)d_in[3];
  const float* wv   = (const float*)d_in[4];
  const float* wc   = (const float*)d_in[5];
  const float* w1w  = (const float*)d_in[6];
  const float* w1b  = (const float*)d_in[7];
  const float* w2w  = (const float*)d_in[8];
  const float* w2b  = (const float*)d_in[9];
  const float* ln1g = (const float*)d_in[10];
  const float* ln1b = (const float*)d_in[11];
  const float* ln2g = (const float*)d_in[12];
  const float* ln2b = (const float*)d_in[13];
  float* out = (float*)d_out;

  constexpr size_t MB = 1024ull * 1024ull;
  char* ws = (char*)d_ws;
  bf16*  xb  = (bf16*)(ws + 0 * MB);     // 16 MB  (8192x1024)
  bf16*  wqb = (bf16*)(ws + 16 * MB);    //  2 MB
  bf16*  wkb = (bf16*)(ws + 18 * MB);    //  2 MB
  bf16*  wvb = (bf16*)(ws + 20 * MB);    //  2 MB
  bf16*  wcb = (bf16*)(ws + 22 * MB);    //  2 MB
  bf16*  w1bm= (bf16*)(ws + 24 * MB);    //  8 MB  (4096x1024)
  bf16*  w2bm= (bf16*)(ws + 32 * MB);    //  8 MB  (1024x4096)
  bf16*  qb  = (bf16*)(ws + 40 * MB);    // 16 MB
  bf16*  kb  = (bf16*)(ws + 56 * MB);    // 16 MB
  bf16*  vb  = (bf16*)(ws + 72 * MB);    // 16 MB
  bf16*  avb = (bf16*)(ws + 88 * MB);    // 16 MB
  float* t1  = (float*)(ws + 104 * MB);  // 32 MB fp32 (reused for z)
  float* u   = (float*)(ws + 136 * MB);  // 32 MB fp32
  bf16*  ub  = (bf16*)(ws + 168 * MB);   // 16 MB
  bf16*  h1b = (bf16*)(ws + 184 * MB);   // 64 MB  (8192x4096)

  auto cvt = [&](const float* in, bf16* o, long n) {
    cvt_bf16_kernel<<<dim3((unsigned)((n + 255) / 256)), dim3(256), 0, stream>>>(in, o, n);
  };
  auto gemm = [&](const bf16* A, const bf16* Bw, const float* bias, float* Cf,
                  bf16* Cb, int Mr, int Nr, int Kd, int relu) {
    int waves = (Mr / 32) * (Nr / 64);
    gemm_wmma_nt<<<dim3(waves / 8), dim3(256), 0, stream>>>(A, Bw, bias, Cf, Cb,
                                                            Mr, Nr, Kd, relu);
  };

  // fp32 -> bf16
  cvt(x,   xb,  (long)NR_ * D_);
  cvt(wq,  wqb, (long)HK_ * D_);
  cvt(wk,  wkb, (long)HK_ * D_);
  cvt(wv,  wvb, (long)HK_ * D_);
  cvt(wc,  wcb, (long)D_ * HK_);
  cvt(w1w, w1bm,(long)M_ * D_);
  cvt(w2w, w2bm,(long)D_ * M_);

  // QKV projections: [8192,1024] = xb @ W^T
  gemm(xb, wqb, nullptr, nullptr, qb, NR_, (int)HK_, D_, 0);
  gemm(xb, wkb, nullptr, nullptr, kb, NR_, (int)HK_, D_, 0);
  gemm(xb, wvb, nullptr, nullptr, vb, NR_, (int)HK_, D_, 0);

  // flash attention: 8192 wave-tiles, 8 waves/block
  flash_attn_kernel<<<dim3(1024), dim3(256), 0, stream>>>(qb, kb, vb, mask, avb);

  // output projection (fp32 accum out) then LN1 with residual x
  gemm(avb, wcb, nullptr, t1, nullptr, NR_, D_, (int)HK_, 0);
  ln_kernel<<<dim3(NR_), dim3(256), 0, stream>>>(t1, x, ln1g, ln1b, u, ub);

  // FFN: h1 = relu(u @ w1^T + b1) [bf16], z = h1 @ w2^T + b2 [fp32]
  gemm(ub,  w1bm, w1b, nullptr, h1b, NR_, M_, D_, 1);
  gemm(h1b, w2bm, w2b, t1, nullptr, NR_, D_, M_, 0);

  // LN2 with residual u -> final fp32 output
  ln_kernel<<<dim3(NR_), dim3(256), 0, stream>>>(t1, u, ln2g, ln2b, out, nullptr);
}
